// MultiHeadAttentionBlock_49357764165751
// MI455X (gfx1250) — matrix operand order repaired and checked by reading, not compiled, not run on hardware
//
#include <hip/hip_runtime.h>
#include <hip/hip_bf16.h>

typedef __attribute__((ext_vector_type(16))) _Float16 v16h;
typedef __attribute__((ext_vector_type(8)))  float    v8f;

#define LN     4096
#define DMODEL 256
#define NH     8
#define DKV    32
#define NG     16

// ---------------------------------------------------------------- helpers

__device__ inline v8f wmma_f16(v16h a, v16h b, v8f c) {
  // v_wmma_f32_16x16x32_f16: (neg_a, A, neg_b, B, c_mod, C, reuse_a, reuse_b)
  return __builtin_amdgcn_wmma_f32_16x16x32_f16(false, a, false, b, (short)0, c,
                                                false, false);
}

// A fragment (16x32 f16, MxK). Lane holds row M = lane&15;
// elem i -> K = hi*16 + i, hi = lane>>4 (matches load_bT_frag).
__device__ inline v16h load_a_frag(const _Float16* __restrict__ base, int ld,
                                   int rowBase, int kBase) {
  const int lane = threadIdx.x & 31;
  const int hi   = lane >> 4;
  const _Float16* p = base + (rowBase + (lane & 15)) * ld + kBase;
  v16h a;
#pragma unroll
  for (int i = 0; i < 16; ++i) a[i] = p[hi * 16 + i];
  return a;
}

// B fragment (32x16 f16, KxN) where B[k][n] = src[(nBase+n)*ld + kBase + k]
// (transpose of a row-major [N][K] source -> contiguous 32-byte loads).
// Lane holds col N = lane&15; elem i -> K = hi*16 + i.
__device__ inline v16h load_bT_frag(const _Float16* __restrict__ src, int ld,
                                    int nBase, int kBase) {
  const int lane = threadIdx.x & 31;
  const int hi   = lane >> 4;
  const _Float16* p = src + (nBase + (lane & 15)) * ld + kBase + hi * 16;
  v16h b;
#pragma unroll
  for (int i = 0; i < 16; ++i) b[i] = p[i];
  return b;
}

// ---------------------------------------------------------------- kernels

__global__ void f32_to_f16_kernel(const float* __restrict__ src,
                                  _Float16* __restrict__ dst, int n) {
  int i = blockIdx.x * blockDim.x + threadIdx.x;
  if (i < n) dst[i] = (_Float16)src[i];
}

// seg[g] = lower_bound(batch, g) for g in [0, NG]; batch is sorted.
__global__ void seg_kernel(const int* __restrict__ batch, int* __restrict__ seg) {
  const int g = threadIdx.x;
  if (g > NG) return;
  int lo = 0, hi = LN;
  while (lo < hi) {
    int mid = (lo + hi) >> 1;
    if (batch[mid] < g) lo = mid + 1; else hi = mid;
  }
  seg[g] = lo;
}

// 16x64 strip of P = x @ W^T + b (then *scale), scattered into the
// "faithful reshape" effective layout.
//   row tile rTile in [0,256): h = rTile>>5, d = rTile&31 are tile-constant;
//   element (rowLocal, c) -> l = rowLocal*256 + c.
// isV==0: dst[h][l][d] (q_eff / k_eff).  isV==1: dst[h][d][l] (v_t).
__global__ void qkv_proj_kernel(const _Float16* __restrict__ x16,
                                const _Float16* __restrict__ w16,
                                const float* __restrict__ bias,
                                _Float16* __restrict__ dst,
                                int isV, float scale) {
  const int cTile = blockIdx.x;   // 0..3  (64-wide strips)
  const int rTile = blockIdx.y;   // 0..255
  const int lane  = threadIdx.x & 31;
  const int hi    = lane >> 4;

  v8f acc[4] = {};
#pragma unroll
  for (int kc = 0; kc < DMODEL; kc += 32) {
    v16h a = load_a_frag(x16, DMODEL, rTile * 16, kc);   // reused 4x
#pragma unroll
    for (int j = 0; j < 4; ++j) {
      v16h b = load_bT_frag(w16, DMODEL, cTile * 64 + j * 16, kc);
      acc[j] = wmma_f16(a, b, acc[j]);
    }
  }

  const int hIdx = rTile >> 5;
  const int dIdx = rTile & 31;
#pragma unroll
  for (int j = 0; j < 4; ++j) {
    const int c    = cTile * 64 + j * 16 + (lane & 15);
    const float bb = bias[c];
#pragma unroll
    for (int v = 0; v < 8; ++v) {
      const int rowLocal = hi * 8 + v;          // C/D layout row
      const float val = (acc[j][v] + bb) * scale;
      const int l = (rowLocal << 8) | c;        // effective token index
      if (isV) dst[(hIdx * DKV + dIdx) * LN + l] = (_Float16)val;
      else     dst[(hIdx * LN  + l)   * DKV + dIdx] = (_Float16)val;
    }
  }
}

// Flash attention: one (head, 32-query tile) per wave32 block.
// batch is sorted -> valid keys for this tile are exactly the contiguous span
// [seg[batch[first_q]], seg[batch[last_q]+1]); masked entries outside contribute 0.
__global__ void attn_kernel(const _Float16* __restrict__ q_eff,
                            const _Float16* __restrict__ k_eff,
                            const _Float16* __restrict__ v_t,
                            const int* __restrict__ batch,
                            const int* __restrict__ seg,
                            _Float16* __restrict__ a16) {
  const int qBlk = blockIdx.x;    // 0..127
  const int h    = blockIdx.y;    // 0..7
  const int lane = threadIdx.x & 31;
  const int hi   = lane >> 4;
  const int nloc = lane & 15;

  __shared__ _Float16 p_sh[32 * 32];
  __shared__ int batch_q[32];

  const _Float16* qh = q_eff + (size_t)h * LN * DKV;
  const _Float16* kh = k_eff + (size_t)h * LN * DKV;
  const _Float16* vh = v_t   + (size_t)h * DKV * LN;

  const int qBase = qBlk * 32;
  batch_q[lane] = batch[qBase + lane];
  __syncthreads();

  // valid key span for this query tile (exact; batch sorted)
  const int gLo = batch[qBase];
  const int gHi = batch[qBase + 31];
  const int mStart = seg[gLo] & ~31;
  int mEnd = (seg[gHi + 1] + 31) & ~31;
  if (mEnd > LN) mEnd = LN;

  v16h aq[2];
  aq[0] = load_a_frag(qh, DKV, qBase, 0);
  aq[1] = load_a_frag(qh, DKV, qBase + 16, 0);

  v8f o0[2] = {}, o1[2] = {};
  float mrow[2][8], lrow[2][8];
#pragma unroll
  for (int t = 0; t < 2; ++t)
#pragma unroll
    for (int v = 0; v < 8; ++v) { mrow[t][v] = -__builtin_inff(); lrow[t][v] = 0.0f; }

  const v8f zero = {};
  for (int mBase = mStart; mBase < mEnd; mBase += 32) {
    if (mBase + 32 < mEnd) {
      __builtin_prefetch(kh + (mBase + 32) * DKV, 0, 1);
      __builtin_prefetch(vh + (mBase + 32), 0, 1);
    }

    // S = Q K^T : B[k=d][n=m] = k_eff[m][d]
    v16h kb0 = load_bT_frag(kh, DKV, mBase,      0);
    v16h kb1 = load_bT_frag(kh, DKV, mBase + 16, 0);
    const int bm0 = batch[mBase + nloc];
    const int bm1 = batch[mBase + nloc + 16];

#pragma unroll
    for (int t = 0; t < 2; ++t) {
      v8f s0 = wmma_f16(aq[t], kb0, zero);
      v8f s1 = wmma_f16(aq[t], kb1, zero);

      // mask + online softmax; row (hi*8+v) lives in the 16 lanes of this half
#pragma unroll
      for (int v = 0; v < 8; ++v) {
        const int bq_ = batch_q[t * 16 + hi * 8 + v];
        const float e0 = (bm0 == bq_) ? (s0[v] + 1.0f) : -__builtin_inff();
        const float e1 = (bm1 == bq_) ? (s1[v] + 1.0f) : -__builtin_inff();
        float cm = fmaxf(e0, e1);
        cm = fmaxf(cm, __shfl_xor(cm, 1, 32));
        cm = fmaxf(cm, __shfl_xor(cm, 2, 32));
        cm = fmaxf(cm, __shfl_xor(cm, 4, 32));
        cm = fmaxf(cm, __shfl_xor(cm, 8, 32));
        const float newm  = fmaxf(mrow[t][v], cm);
        const float alpha =
            (newm == -__builtin_inff()) ? 1.0f : __expf(mrow[t][v] - newm);
        const float p0 = (e0 == -__builtin_inff()) ? 0.0f : __expf(e0 - newm);
        const float p1 = (e1 == -__builtin_inff()) ? 0.0f : __expf(e1 - newm);
        float rs = p0 + p1;
        rs += __shfl_xor(rs, 1, 32);
        rs += __shfl_xor(rs, 2, 32);
        rs += __shfl_xor(rs, 4, 32);
        rs += __shfl_xor(rs, 8, 32);
        lrow[t][v] = lrow[t][v] * alpha + rs;
        mrow[t][v] = newm;
        o0[t][v] *= alpha;
        o1[t][v] *= alpha;
        const int row = t * 16 + hi * 8 + v;     // C-layout -> LDS, row-major
        p_sh[row * 32 + nloc]      = (_Float16)p0;
        p_sh[row * 32 + nloc + 16] = (_Float16)p1;
      }
    }
    __syncthreads();

    // O += P V : A = P (16x32 via LDS transpose), B[k=m][n=d] = v_t[h][d][m]
    v16h vb0 = load_bT_frag(vh, LN, 0,  mBase);
    v16h vb1 = load_bT_frag(vh, LN, 16, mBase);
#pragma unroll
    for (int t = 0; t < 2; ++t) {
      v16h ap = load_a_frag(p_sh, 32, t * 16, 0);
      o0[t] = wmma_f16(ap, vb0, o0[t]);
      o1[t] = wmma_f16(ap, vb1, o1[t]);
    }
    __syncthreads();
  }

  // normalize + scatter into faithful a.reshape(L, 256) layout, f16 for Wo GEMM
#pragma unroll
  for (int t = 0; t < 2; ++t)
#pragma unroll
    for (int v = 0; v < 8; ++v) {
      const int l = qBase + t * 16 + hi * 8 + v;
      const float inv = 1.0f / lrow[t][v];
      const int aRow = h * 512 + (l >> 3);
      const int aCol = (l & 7) * 32;
      a16[aRow * DMODEL + aCol + nloc]      = (_Float16)(o0[t][v] * inv);
      a16[aRow * DMODEL + aCol + nloc + 16] = (_Float16)(o1[t][v] * inv);
    }
}

// out = a @ Wo^T + bo, f32 output, 16x64 strip per wave
__global__ void out_proj_kernel(const _Float16* __restrict__ a16,
                                const _Float16* __restrict__ wo16,
                                const float* __restrict__ bo,
                                float* __restrict__ out) {
  const int cTile = blockIdx.x;   // 0..3
  const int rTile = blockIdx.y;   // 0..255
  const int lane  = threadIdx.x & 31;
  const int hi    = lane >> 4;

  v8f acc[4] = {};
#pragma unroll
  for (int kc = 0; kc < DMODEL; kc += 32) {
    v16h a = load_a_frag(a16, DMODEL, rTile * 16, kc);
#pragma unroll
    for (int j = 0; j < 4; ++j) {
      v16h b = load_bT_frag(wo16, DMODEL, cTile * 64 + j * 16, kc);
      acc[j] = wmma_f16(a, b, acc[j]);
    }
  }
#pragma unroll
  for (int j = 0; j < 4; ++j) {
    const int c = cTile * 64 + j * 16 + (lane & 15);
    const float bb = bo[c];
#pragma unroll
    for (int v = 0; v < 8; ++v) {
      const int r = rTile * 16 + hi * 8 + v;
      out[r * DMODEL + c] = acc[j][v] + bb;
    }
  }
}

// ---------------------------------------------------------------- launch

extern "C" void kernel_launch(void* const* d_in, const int* in_sizes, int n_in,
                              void* d_out, int out_size, void* d_ws, size_t ws_size,
                              hipStream_t stream) {
  const float* x     = (const float*)d_in[0];
  const int*   batch = (const int*)d_in[1];
  const float* Wq    = (const float*)d_in[2];
  const float* bq    = (const float*)d_in[3];
  const float* Wk    = (const float*)d_in[4];
  const float* bk    = (const float*)d_in[5];
  const float* Wv    = (const float*)d_in[6];
  const float* bv    = (const float*)d_in[7];
  const float* Wo    = (const float*)d_in[8];
  const float* bo    = (const float*)d_in[9];
  float* out = (float*)d_out;

  // workspace carve-up (f16 first, then int seg), ~10.5 MB total
  _Float16* w = (_Float16*)d_ws;
  _Float16* x16   = w; w += (size_t)LN * DMODEL;
  _Float16* wq16  = w; w += (size_t)DMODEL * DMODEL;
  _Float16* wk16  = w; w += (size_t)DMODEL * DMODEL;
  _Float16* wv16  = w; w += (size_t)DMODEL * DMODEL;
  _Float16* wo16  = w; w += (size_t)DMODEL * DMODEL;
  _Float16* q_eff = w; w += (size_t)NH * LN * DKV;
  _Float16* k_eff = w; w += (size_t)NH * LN * DKV;
  _Float16* v_t   = w; w += (size_t)NH * DKV * LN;
  _Float16* a16   = w; w += (size_t)LN * DMODEL;
  int* seg = (int*)w;                       // 17 ints, 4-byte aligned offset

  const int nx = LN * DMODEL;
  const int nw = DMODEL * DMODEL;
  f32_to_f16_kernel<<<(nx + 255) / 256, 256, 0, stream>>>(x,  x16,  nx);
  f32_to_f16_kernel<<<(nw + 255) / 256, 256, 0, stream>>>(Wq, wq16, nw);
  f32_to_f16_kernel<<<(nw + 255) / 256, 256, 0, stream>>>(Wk, wk16, nw);
  f32_to_f16_kernel<<<(nw + 255) / 256, 256, 0, stream>>>(Wv, wv16, nw);
  f32_to_f16_kernel<<<(nw + 255) / 256, 256, 0, stream>>>(Wo, wo16, nw);
  seg_kernel<<<1, 32, 0, stream>>>(batch, seg);

  const float scale = 0.0625f;  // 1/sqrt(256)
  dim3 gGemm(DMODEL / 64, LN / 16);   // (4, 256)
  qkv_proj_kernel<<<gGemm, 32, 0, stream>>>(x16, wq16, bq, q_eff, 0, scale);
  qkv_proj_kernel<<<gGemm, 32, 0, stream>>>(x16, wk16, bk, k_eff, 0, scale);
  qkv_proj_kernel<<<gGemm, 32, 0, stream>>>(x16, wv16, bv, v_t,   1, 1.0f);

  dim3 gAttn(LN / 32, NH);            // (128, 8)
  attn_kernel<<<gAttn, 32, 0, stream>>>(q_eff, k_eff, v_t, batch, seg, a16);

  out_proj_kernel<<<gGemm, 32, 0, stream>>>(a16, wo16, bo, out);
}
